// UnifiedContrastiveModel_31559419691045
// MI455X (gfx1250) — compile-verified, matrix operand order checked by reading, not run on hardware
//
#include <hip/hip_runtime.h>
#include <hip/hip_bf16.h>

// ---------------- problem constants (from reference) ----------------
#define Bn   8192      // nodes / batch
#define Cc   256       // channels
#define Ll   80        // seq len
#define OUTC 64        // out_channels
#define HIDC 128       // hidden
#define FDC  128       // feature_dim
#define NWw  15        // windows
#define NTt  (Bn*NWw)  // temporal nodes = 122880
#define E_S  262144
#define E_T  983040

typedef __attribute__((ext_vector_type(16))) _Float16 v16h;
typedef __attribute__((ext_vector_type(8)))  float    v8f;

// K-index inside a 16x32 f16 A fragment (ISA 7.12.2, wave32):
// VGPR v in 0..7 holds an element pair; lanes split K by half-wave.
__device__ __forceinline__ int frag_kb(int v, int half) {
  return (v < 4) ? (half * 8 + 2 * v) : (16 + half * 8 + 2 * (v - 4));
}

// ---------------------------------------------------------------
// Fused streaming pass over x: node_x = mean over L, red = 15
// sliding-window means (win=10, stride=5) via 16 pentad partials.
// ---------------------------------------------------------------
__global__ __launch_bounds__(256) void k_window_reduce(
    const float* __restrict__ x, float* __restrict__ node_x, float* __restrict__ red) {
  int r = blockIdx.x * blockDim.x + threadIdx.x;
  if (r >= Bn * Cc) return;
  const float* p = x + (size_t)r * Ll;
  float pent[16];
  float tot = 0.f;
#pragma unroll
  for (int j = 0; j < 16; ++j) {
    float s = 0.f;
#pragma unroll
    for (int i = 0; i < 5; ++i) s += p[j * 5 + i];
    pent[j] = s;
    tot += s;
  }
  node_x[r] = tot * (1.0f / 80.0f);
  int b = r / Cc, c = r % Cc;
#pragma unroll
  for (int w = 0; w < NWw; ++w)
    red[((size_t)b * NWw + w) * Cc + c] = (pent[w] + pent[w + 1]) * 0.1f;
}

// ---------------------------------------------------------------
// Pack fp32 weight (K x M row-major) into f16 WMMA B-fragments:
// pw[((tm*KS + ks)*32 + lane)*16 + e]  -> per-lane 32B contiguous,
// consecutive lanes contiguous. One thread per (tm, ks, lane).
// ---------------------------------------------------------------
__global__ __launch_bounds__(256) void k_pack_w(
    const float* __restrict__ W, _Float16* __restrict__ pw, int K, int M) {
  int t = blockIdx.x * blockDim.x + threadIdx.x;
  int KS = K >> 5, MT = M >> 4;
  if (t >= MT * KS * 32) return;
  int lane = t & 31;
  int rest = t >> 5;
  int ks = rest % KS, tm = rest / KS;
  int r16 = lane & 15, half = lane >> 4;
  int col = tm * 16 + r16;
  v16h f;
#pragma unroll
  for (int v = 0; v < 8; ++v) {
    int k = ks * 32 + frag_kb(v, half);
    f[2 * v]     = (_Float16)W[(size_t)k * M + col];
    f[2 * v + 1] = (_Float16)W[(size_t)(k + 1) * M + col];
  }
  *(v16h*)(pw + (size_t)t * 16) = f;
}

// ---------------- per-destination edge counts -------------------
__global__ __launch_bounds__(256) void k_count(
    const int* __restrict__ dst, int E, float* __restrict__ cnt) {
  int e = blockIdx.x * blockDim.x + threadIdx.x;
  if (e < E) atomicAdd(&cnt[dst[e]], 1.0f);
}

// ---------------- scatter-add of features over edges ------------
__global__ __launch_bounds__(256) void k_scatter(
    const int* __restrict__ src, const int* __restrict__ dst, int E,
    const float* __restrict__ X, int C, float* __restrict__ agg) {
  unsigned t = blockIdx.x * blockDim.x + threadIdx.x;
  unsigned c4 = (unsigned)(C >> 2);
  if (t >= (unsigned)E * c4) return;
  unsigned e = t / c4;
  unsigned c = (t - e * c4) * 4u;
  int s = src[e], d = dst[e];
  const float4 v = *(const float4*)(X + (size_t)s * C + c);
  float* p = agg + (size_t)d * C + c;
  atomicAdd(p + 0, v.x);
  atomicAdd(p + 1, v.y);
  atomicAdd(p + 2, v.z);
  atomicAdd(p + 3, v.w);
}

// ---------------- mean = agg / max(cnt,1) -----------------------
__global__ __launch_bounds__(256) void k_mean_div(
    float* __restrict__ agg, const float* __restrict__ cnt, int N, int C) {
  int t = blockIdx.x * blockDim.x + threadIdx.x;
  if (t >= N * C) return;
  int n = t / C;
  agg[t] /= fmaxf(cnt[n], 1.0f);
}

// ---------------------------------------------------------------
// Strip-mined WMMA GEMM with LDS-staged weights.
// One wave owns a 16 x M output strip; packed weight fragments are
// cooperatively staged into LDS once per block (max 64KB), then each
// wave feeds WMMA B operands from LDS (ds_load_b128 x2 / fragment).
// Dual-input layers reuse the same LDS buffer: stage W1 -> pass 1 ->
// barrier -> stage W2 -> pass 2.
// REQUIREMENT: N % 128 == 0 (full blocks; uniform barriers, EXEC=~0).
// ---------------------------------------------------------------
template <int K, int M>
__device__ __forceinline__ void stage_lds(
    const _Float16* __restrict__ PW, _Float16* sm, int tid) {
  constexpr int VECS = (K * M) / 8;  // uint4 = 8 halves = 16B
  const uint4* src = (const uint4*)PW;
  uint4* dst = (uint4*)sm;
#pragma unroll
  for (int i = tid; i < VECS; i += 256) dst[i] = src[i];
}

template <int K, int M>
__device__ __forceinline__ void strip_pass(
    const float* __restrict__ A, const _Float16* sm,
    v8f* acc, int arow, int half, int lane) {
  constexpr int KS = K / 32, MT = M / 16;
#pragma unroll
  for (int ks = 0; ks < KS; ++ks) {
    v16h a;
#pragma unroll
    for (int v = 0; v < 8; ++v) {
      int k = ks * 32 + frag_kb(v, half);
      a[2 * v]     = (_Float16)A[(size_t)arow * K + k];
      a[2 * v + 1] = (_Float16)A[(size_t)arow * K + k + 1];
    }
#pragma unroll
    for (int tm = 0; tm < MT; ++tm) {
      const v16h b = *(const v16h*)(sm + ((tm * KS + ks) * 32 + lane) * 16);
      acc[tm] = __builtin_amdgcn_wmma_f32_16x16x32_f16(
          /*neg_a=*/false, a, /*neg_b=*/false, b,
          /*c_mod=*/(short)0, acc[tm], /*reuse_a=*/false, /*reuse_b=*/false);
    }
  }
}

template <int K, int M, bool DUAL, bool RELU>
__global__ __launch_bounds__(256) void k_gemm_wmma(
    const float* __restrict__ A1, const float* __restrict__ A2,
    const _Float16* __restrict__ PW1, const _Float16* __restrict__ PW2,
    const float* __restrict__ bias, float* __restrict__ out, int N) {
  constexpr int MT = M / 16;
  __shared__ _Float16 smw[K * M];          // <= 64KB (K=256,M=128)
  int tid  = threadIdx.x;
  int wid  = tid >> 5;
  int lane = tid & 31;
  int strip = blockIdx.x * 8 + wid;        // grid sized exactly: no tail waves
  int r16 = lane & 15, half = lane >> 4;
  int arow = strip * 16 + r16;

  stage_lds<K, M>(PW1, smw, tid);
  __syncthreads();

  v8f acc[MT];
#pragma unroll
  for (int tm = 0; tm < MT; ++tm) acc[tm] = {};

  strip_pass<K, M>(A1, smw, acc, arow, half, lane);
  if (DUAL) {
    __syncthreads();                        // all waves done reading W1
    stage_lds<K, M>(PW2, smw, tid);
    __syncthreads();
    strip_pass<K, M>(A2, smw, acc, arow, half, lane);
  }

#pragma unroll
  for (int tm = 0; tm < MT; ++tm) {
    int ocol = tm * 16 + r16;
    float bv = bias ? bias[ocol] : 0.0f;
#pragma unroll
    for (int i = 0; i < 8; ++i) {
      int orow = strip * 16 + i + 8 * half;
      float v = acc[tm][i] + bv;
      if (RELU) v = fmaxf(v, 0.0f);
      out[(size_t)orow * M + ocol] = v;
    }
  }
  (void)N;
}

// ---------------- row-wise L2 normalize -------------------------
__global__ __launch_bounds__(256) void k_l2norm(float* __restrict__ X, int N, int M) {
  int r = blockIdx.x * blockDim.x + threadIdx.x;
  if (r >= N) return;
  float* p = X + (size_t)r * M;
  float ss = 0.f;
  for (int c = 0; c < M; ++c) { float v = p[c]; ss += v * v; }
  float inv = 1.0f / fmaxf(sqrtf(ss), 1e-12f);
  for (int c = 0; c < M; ++c) p[c] *= inv;
}

// ---------------- mean over NW windows -------------------------
__global__ __launch_bounds__(256) void k_pool(
    const float* __restrict__ t_emb, float* __restrict__ out) {
  int t = blockIdx.x * blockDim.x + threadIdx.x;
  if (t >= Bn * OUTC) return;
  int b = t / OUTC, c = t - b * OUTC;
  float s = 0.f;
#pragma unroll
  for (int w = 0; w < NWw; ++w) s += t_emb[((size_t)b * NWw + w) * OUTC + c];
  out[t] = s * (1.0f / NWw);
}

// ===============================================================
extern "C" void kernel_launch(void* const* d_in, const int* in_sizes, int n_in,
                              void* d_out, int out_size, void* d_ws, size_t ws_size,
                              hipStream_t stream) {
  const float* x   = (const float*)d_in[0];
  const int*   sei = (const int*)d_in[1];   // [2, E_S]: row0=src, row1=dst
  const int*   tei = (const int*)d_in[2];   // [2, E_T]
  const float* s_l1_W  = (const float*)d_in[3];
  const float* s_l1_b  = (const float*)d_in[4];
  const float* s_l1_Wr = (const float*)d_in[5];
  const float* s_l2_W  = (const float*)d_in[6];
  const float* s_l2_b  = (const float*)d_in[7];
  const float* s_l2_Wr = (const float*)d_in[8];
  const float* s_p1_W  = (const float*)d_in[9];
  const float* s_p1_b  = (const float*)d_in[10];
  const float* s_p2_W  = (const float*)d_in[11];
  const float* s_p2_b  = (const float*)d_in[12];
  const float* t_l1_W  = (const float*)d_in[13];
  const float* t_l1_b  = (const float*)d_in[14];
  const float* t_l1_Wr = (const float*)d_in[15];
  const float* t_l2_W  = (const float*)d_in[16];
  const float* t_l2_b  = (const float*)d_in[17];
  const float* t_l2_Wr = (const float*)d_in[18];
  const float* t_p1_W  = (const float*)d_in[19];
  const float* t_p1_b  = (const float*)d_in[20];
  const float* t_p2_W  = (const float*)d_in[21];
  const float* t_p2_b  = (const float*)d_in[22];

  // -------- workspace carve-out (256B aligned) --------
  char* ws = (char*)d_ws;
  size_t off = 0;
  auto carve = [&](size_t bytes) -> char* {
    char* p = ws + off;
    off += (bytes + 255) & ~(size_t)255;
    return p;
  };
  float* node_x = (float*)carve((size_t)Bn * Cc * 4);
  float* red    = (float*)carve((size_t)NTt * Cc * 4);
  float* agg    = (float*)carve((size_t)NTt * Cc * 4);   // reused by all 4 scatter phases
  float* s_cnt  = (float*)carve((size_t)Bn * 4);
  float* t_cnt  = (float*)carve((size_t)NTt * 4);
  float* s_h1   = (float*)carve((size_t)Bn * HIDC * 4);
  float* s_emb  = (float*)carve((size_t)Bn * OUTC * 4);
  float* s_hid  = (float*)carve((size_t)Bn * FDC * 4);
  float* t_h1   = (float*)carve((size_t)NTt * HIDC * 4);
  float* t_emb  = (float*)carve((size_t)NTt * OUTC * 4);
  float* t_pool = (float*)carve((size_t)Bn * OUTC * 4);
  float* t_hid  = (float*)carve((size_t)Bn * FDC * 4);

  // packed f16 weight fragments
  struct WPack { const float* src; int K, M; _Float16* dst; };
  WPack wp[12] = {
    {s_l1_W,  Cc,   HIDC, nullptr}, {s_l1_Wr, Cc,   HIDC, nullptr},
    {s_l2_W,  HIDC, OUTC, nullptr}, {s_l2_Wr, HIDC, OUTC, nullptr},
    {s_p1_W,  OUTC, FDC,  nullptr}, {s_p2_W,  FDC,  FDC,  nullptr},
    {t_l1_W,  Cc,   HIDC, nullptr}, {t_l1_Wr, Cc,   HIDC, nullptr},
    {t_l2_W,  HIDC, OUTC, nullptr}, {t_l2_Wr, HIDC, OUTC, nullptr},
    {t_p1_W,  OUTC, FDC,  nullptr}, {t_p2_W,  FDC,  FDC,  nullptr},
  };
  for (int i = 0; i < 12; ++i) wp[i].dst = (_Float16*)carve((size_t)wp[i].K * wp[i].M * 2);
  for (int i = 0; i < 12; ++i) {
    int thr = (wp[i].M / 16) * (wp[i].K / 32) * 32;
    k_pack_w<<<(thr + 255) / 256, 256, 0, stream>>>(wp[i].src, wp[i].dst, wp[i].K, wp[i].M);
  }
  _Float16 *ps_l1W = wp[0].dst, *ps_l1Wr = wp[1].dst, *ps_l2W = wp[2].dst, *ps_l2Wr = wp[3].dst,
           *ps_p1W = wp[4].dst, *ps_p2W = wp[5].dst;
  _Float16 *pt_l1W = wp[6].dst, *pt_l1Wr = wp[7].dst, *pt_l2W = wp[8].dst, *pt_l2Wr = wp[9].dst,
           *pt_p1W = wp[10].dst, *pt_p2W = wp[11].dst;

  float* s_proj = (float*)d_out;                       // [8192,128]
  float* t_proj = (float*)d_out + (size_t)Bn * FDC;    // [8192,128]

  // exact grids: Bn=8192 and NTt=122880 are both multiples of 16*8=128
  auto strips = [](int N) { return (N >> 4) / 8; };

  // -------- fused windows + node mean (single pass over x) --------
  k_window_reduce<<<(Bn * Cc + 255) / 256, 256, 0, stream>>>(x, node_x, red);

  // ===================== SPATIAL branch =====================
  hipMemsetAsync(s_cnt, 0, (size_t)Bn * 4, stream);
  hipMemsetAsync(agg, 0, (size_t)Bn * Cc * 4, stream);
  k_count<<<(E_S + 255) / 256, 256, 0, stream>>>(sei + E_S, E_S, s_cnt);
  {
    unsigned nt = (unsigned)E_S * (Cc / 4);
    k_scatter<<<(nt + 255) / 256, 256, 0, stream>>>(sei, sei + E_S, E_S, node_x, Cc, agg);
  }
  k_mean_div<<<(Bn * Cc + 255) / 256, 256, 0, stream>>>(agg, s_cnt, Bn, Cc);
  k_gemm_wmma<Cc, HIDC, true, true><<<strips(Bn), 256, 0, stream>>>(
      agg, node_x, ps_l1W, ps_l1Wr, s_l1_b, s_h1, Bn);

  hipMemsetAsync(agg, 0, (size_t)Bn * HIDC * 4, stream);
  {
    unsigned nt = (unsigned)E_S * (HIDC / 4);
    k_scatter<<<(nt + 255) / 256, 256, 0, stream>>>(sei, sei + E_S, E_S, s_h1, HIDC, agg);
  }
  k_mean_div<<<(Bn * HIDC + 255) / 256, 256, 0, stream>>>(agg, s_cnt, Bn, HIDC);
  k_gemm_wmma<HIDC, OUTC, true, false><<<strips(Bn), 256, 0, stream>>>(
      agg, s_h1, ps_l2W, ps_l2Wr, s_l2_b, s_emb, Bn);
  k_l2norm<<<(Bn + 255) / 256, 256, 0, stream>>>(s_emb, Bn, OUTC);

  k_gemm_wmma<OUTC, FDC, false, true><<<strips(Bn), 256, 0, stream>>>(
      s_emb, nullptr, ps_p1W, nullptr, s_p1_b, s_hid, Bn);
  k_gemm_wmma<FDC, FDC, false, false><<<strips(Bn), 256, 0, stream>>>(
      s_hid, nullptr, ps_p2W, nullptr, s_p2_b, s_proj, Bn);
  k_l2norm<<<(Bn + 255) / 256, 256, 0, stream>>>(s_proj, Bn, FDC);

  // ===================== TEMPORAL branch =====================
  hipMemsetAsync(t_cnt, 0, (size_t)NTt * 4, stream);
  hipMemsetAsync(agg, 0, (size_t)NTt * Cc * 4, stream);
  k_count<<<(E_T + 255) / 256, 256, 0, stream>>>(tei + E_T, E_T, t_cnt);
  {
    unsigned nt = (unsigned)E_T * (Cc / 4);
    k_scatter<<<(nt + 255) / 256, 256, 0, stream>>>(tei, tei + E_T, E_T, red, Cc, agg);
  }
  k_mean_div<<<(NTt * Cc + 255) / 256, 256, 0, stream>>>(agg, t_cnt, NTt, Cc);
  k_gemm_wmma<Cc, HIDC, true, true><<<strips(NTt), 256, 0, stream>>>(
      agg, red, pt_l1W, pt_l1Wr, t_l1_b, t_h1, NTt);

  hipMemsetAsync(agg, 0, (size_t)NTt * HIDC * 4, stream);
  {
    unsigned nt = (unsigned)E_T * (HIDC / 4);
    k_scatter<<<(nt + 255) / 256, 256, 0, stream>>>(tei, tei + E_T, E_T, t_h1, HIDC, agg);
  }
  k_mean_div<<<(NTt * HIDC + 255) / 256, 256, 0, stream>>>(agg, t_cnt, NTt, HIDC);
  k_gemm_wmma<HIDC, OUTC, true, false><<<strips(NTt), 256, 0, stream>>>(
      agg, t_h1, pt_l2W, pt_l2Wr, t_l2_b, t_emb, NTt);
  k_l2norm<<<(NTt + 255) / 256, 256, 0, stream>>>(t_emb, NTt, OUTC);

  k_pool<<<(Bn * OUTC + 255) / 256, 256, 0, stream>>>(t_emb, t_pool);
  k_gemm_wmma<OUTC, FDC, false, true><<<strips(Bn), 256, 0, stream>>>(
      t_pool, nullptr, pt_p1W, nullptr, t_p1_b, t_hid, Bn);
  k_gemm_wmma<FDC, FDC, false, false><<<strips(Bn), 256, 0, stream>>>(
      t_hid, nullptr, pt_p2W, nullptr, t_p2_b, t_proj, Bn);
  k_l2norm<<<(Bn + 255) / 256, 256, 0, stream>>>(t_proj, Bn, FDC);

  (void)in_sizes; (void)n_in; (void)out_size; (void)ws_size;
}
